// sLSTMblock_32676111188021
// MI455X (gfx1250) — matrix operand order repaired and checked by reading, not compiled, not run on hardware
//
#include <hip/hip_runtime.h>
#include <math.h>

#define E 512
#define KCONV 64
#define BT 4096           // B*T = 4*1024
#define T_LEN 1024
#define E43 682
#define EPS 1e-5f
#define NCHUNK 1024       // (E*KCONV)/32

typedef __attribute__((ext_vector_type(16))) __bf16 v16bf;
typedef __attribute__((ext_vector_type(8)))  __bf16 v8bf;
typedef __attribute__((ext_vector_type(8)))  float  v8f;
typedef __attribute__((ext_vector_type(4)))  int    v4i;

// ---- gfx1250 async global->LDS staging (guarded; fallback = sync LDS copy) ----
#if defined(__gfx1250__) && __has_builtin(__builtin_amdgcn_global_load_async_to_lds_b128)
#define HAVE_ASYNC_LDS 1
#define ASYNC_GPTR(p) ((__attribute__((address_space(1))) v4i*)(p))
#define ASYNC_LPTR(p) ((__attribute__((address_space(3))) v4i*)(p))
#endif

#if defined(__AMDGCN__)
  #if __has_builtin(__builtin_amdgcn_s_wait_asynccnt)
    #define WAIT_ASYNC() __builtin_amdgcn_s_wait_asynccnt(0)
  #else
    #define WAIT_ASYNC() asm volatile("s_wait_asynccnt 0x0" ::: "memory")
  #endif
#else
  #define WAIT_ASYNC()
#endif

// ---------------- block reduction helpers ----------------
__device__ __forceinline__ float block_sum(float v, float* buf) {
  const int t = threadIdx.x;
  buf[t] = v;
  __syncthreads();
  for (int s = blockDim.x >> 1; s > 0; s >>= 1) {
    if (t < s) buf[t] += buf[t + s];
    __syncthreads();
  }
  float r = buf[0];
  __syncthreads();
  return r;
}

__device__ __forceinline__ float ln_val(float v, int n, bool active,
                                        float g, float b, float* buf) {
  float x = active ? v : 0.f;
  float s1 = block_sum(x, buf);
  float s2 = block_sum(x * x, buf);
  float mu = s1 / (float)n;
  float var = s2 / (float)n - mu * mu;
  return (v - mu) * rsqrtf(var + EPS) * g + b;
}

// ---------------- K0: zero accumulators ----------------
__global__ void k_zero(float* p, int n) {
  int i = blockIdx.x * blockDim.x + threadIdx.x;
  if (i < n) p[i] = 0.f;
}

// ---------------- K1: LN(x) -> f32 + bf16 copies ----------------
__global__ void k_ln_x(const float* __restrict__ x,
                       const float* __restrict__ g, const float* __restrict__ b,
                       float* __restrict__ xln, __bf16* __restrict__ xbf) {
  __shared__ float buf[E];
  const int row = blockIdx.x;
  const int e = threadIdx.x;
  float v = x[(size_t)row * E + e];
  float y = ln_val(v, E, true, g[e], b[e], buf);
  xln[(size_t)row * E + e] = y;
  xbf[(size_t)row * E + e] = (__bf16)y;
}

// ---------------- K2: repack conv_w [O,I,K] -> bf16 chunk-major ----------------
// Wpack[((c*E)+o)*32 + j] = conv_w[o, (c%16)*32 + j, c/16]
__global__ void k_pack_w(const float* __restrict__ w, __bf16* __restrict__ wp) {
  const size_t total = (size_t)NCHUNK * E * 32;
  for (size_t idx = (size_t)blockIdx.x * blockDim.x + threadIdx.x; idx < total;
       idx += (size_t)gridDim.x * blockDim.x) {
    int j = (int)(idx & 31);
    int o = (int)((idx >> 5) & (E - 1));
    int c = (int)(idx >> 14);
    int k = c >> 4;
    int i = ((c & 15) << 5) + j;
    wp[idx] = (__bf16)w[((size_t)o * E + i) * KCONV + k];
  }
}

// ---------------- K2b: recurrent contributions r = bd(ht1, rw) ----------------
__global__ void k_recur(const float* __restrict__ h,
                        const float* __restrict__ rwi, const float* __restrict__ rwf,
                        const float* __restrict__ rwo, const float* __restrict__ rwz,
                        float* __restrict__ ri, float* __restrict__ rf,
                        float* __restrict__ ro, float* __restrict__ rz) {
  const int e = threadIdx.x;
  const int d = e >> 6, o = e & 63;
  const int base = (d * 64 + o) * 64;
  float si = 0.f, sf = 0.f, so = 0.f, sz = 0.f;
#pragma unroll 8
  for (int j = 0; j < 64; ++j) {
    float hv = h[d * 64 + j];
    si += hv * rwi[base + j];
    sf += hv * rwf[base + j];
    so += hv * rwo[base + j];
    sz += hv * rwz[base + j];
  }
  ri[e] = si; rf[e] = sf; ro[e] = so; rz[e] = sz;
}

// ---------------- K3: causal conv as implicit GEMM via WMMA bf16 ----------------
// grid (64, 2), block 256 (8 waves as 2 Mx x 4 Nx). Wave: 32 rows x 64 cols.
// Weight chunks staged global->LDS with async-to-LDS, double buffered.

__device__ __forceinline__ v16bf load_afrag(const __bf16* __restrict__ xbf,
                                            int row, int k, int i0, int asub) {
  v16bf a;
  if (((row & (T_LEN - 1)) + k) >= (KCONV - 1)) {
    const __bf16* ap = xbf + (size_t)(row + k - (KCONV - 1)) * E + i0 + asub;
    v8bf lo = *(const v8bf*)(ap);        // K = sub+0..7
    v8bf hi = *(const v8bf*)(ap + 16);   // K = sub+16..23
#pragma unroll
    for (int e2 = 0; e2 < 8; ++e2) { a[e2] = lo[e2]; a[8 + e2] = hi[e2]; }
  } else {
#pragma unroll
    for (int e2 = 0; e2 < 16; ++e2) a[e2] = (__bf16)0.f;
  }
  return a;
}

__device__ __forceinline__ v16bf load_bfrag_lds(const __bf16* lb, int o_local, int bsub) {
  const __bf16* bp = lb + (size_t)o_local * 32 + bsub;
  v8bf lo = *(const v8bf*)(bp);      // ds_load_b128
  v8bf hi = *(const v8bf*)(bp + 16);
  v16bf b;
#pragma unroll
  for (int e2 = 0; e2 < 8; ++e2) { b[e2] = lo[e2]; b[8 + e2] = hi[e2]; }
  return b;
}

// stage one 16KB weight chunk window (256 cols x 32 K, bf16) into LDS
__device__ __forceinline__ void stage_chunk(const __bf16* __restrict__ wp,
                                            int c, int nb, __bf16* dst, int tid) {
  const char* g = (const char*)(wp + ((size_t)c * E + nb) * 32) + tid * 64;
  char* l = (char*)dst + tid * 64;
#if defined(HAVE_ASYNC_LDS)
  __builtin_amdgcn_global_load_async_to_lds_b128(ASYNC_GPTR(g),      ASYNC_LPTR(l),      0, 0);
  __builtin_amdgcn_global_load_async_to_lds_b128(ASYNC_GPTR(g + 16), ASYNC_LPTR(l + 16), 0, 0);
  __builtin_amdgcn_global_load_async_to_lds_b128(ASYNC_GPTR(g + 32), ASYNC_LPTR(l + 32), 0, 0);
  __builtin_amdgcn_global_load_async_to_lds_b128(ASYNC_GPTR(g + 48), ASYNC_LPTR(l + 48), 0, 0);
#else
#pragma unroll
  for (int i = 0; i < 4; ++i)
    *(v8bf*)(l + 16 * i) = *(const v8bf*)(g + 16 * i);
#endif
}

__global__ __launch_bounds__(256) void k_conv_wmma(const __bf16* __restrict__ xbf,
                                                   const __bf16* __restrict__ wp,
                                                   const float* __restrict__ conv_b,
                                                   float* __restrict__ xconv) {
  __shared__ __bf16 lb[2][256 * 32];   // double-buffered weight chunk (2 x 16KB)

  const int tid = threadIdx.x;
  const int lane = tid & 31;
  const int wave = tid >> 5;
  const int msub = wave >> 2;                       // 0..1
  const int nsub = wave & 3;                        // 0..3
  const int m0 = (blockIdx.x << 6) + (msub << 5);   // wave's 32 rows
  const int nb = blockIdx.y << 8;                   // block's 256-col window

  const int asub = (lane >= 16) ? 8 : 0;
  const int bsub = (lane >= 16) ? 8 : 0;
  const int ar0 = m0 + (lane & 15);
  const int ar1 = ar0 + 16;
  const int ncol_l = (nsub << 6) + (lane & 15);     // frag col base in window

  v8f acc[8];
#pragma unroll
  for (int i = 0; i < 8; ++i) acc[i] = (v8f){};

  // prologue: stage chunk 0
  stage_chunk(wp, 0, nb, lb[0], tid);
  WAIT_ASYNC();
  __syncthreads();

  for (int c = 0; c < NCHUNK; ++c) {
    const int k = c >> 4;
    const int i0 = (c & 15) << 5;
    const __bf16* cur = lb[c & 1];

    // overlap: issue async loads for next chunk into the other buffer
    if (c + 1 < NCHUNK) stage_chunk(wp, c + 1, nb, lb[(c + 1) & 1], tid);

    // prefetch A rows for the next k-step (tiny, L2-resident stream)
    if (c + 16 < NCHUNK)
      __builtin_prefetch(xbf + (size_t)(ar0 + (k + 1) - (KCONV - 1)) * E, 0, 1);

    const v16bf a0 = load_afrag(xbf, ar0, k, i0, asub);
    const v16bf a1 = load_afrag(xbf, ar1, k, i0, asub);

#pragma unroll
    for (int nt = 0; nt < 4; ++nt) {
      const v16bf bfrag = load_bfrag_lds(cur, ncol_l + nt * 16, bsub);
      acc[nt] = __builtin_amdgcn_wmma_f32_16x16x32_bf16(false, a0, false, bfrag,
                                                        (short)0, acc[nt], false, false);
      acc[4 + nt] = __builtin_amdgcn_wmma_f32_16x16x32_bf16(false, a1, false, bfrag,
                                                            (short)0, acc[4 + nt], false, false);
    }

    WAIT_ASYNC();      // next buffer landed
    __syncthreads();   // all waves done reading cur & see next buffer
  }

  // ---- epilogue: bias + SiLU, store x_conv ----
#pragma unroll
  for (int mi = 0; mi < 2; ++mi) {
#pragma unroll
    for (int nt = 0; nt < 4; ++nt) {
      const int col = nb + ncol_l + nt * 16;
      const float bias = conv_b[col];
#pragma unroll
      for (int v = 0; v < 8; ++v) {
        const int row = m0 + mi * 16 + ((lane < 16) ? v : v + 8);
        float val = acc[mi * 4 + nt][v] + bias;
        const float sig = 1.f / (1.f + __expf(-val));
        xconv[(size_t)row * E + col] = val * sig;
      }
    }
  }
}

// ---------------- K4: gates + row LNs + (B,T)-mean accumulation ----------------
__global__ void k_gates(const float* __restrict__ xconv, const float* __restrict__ xln,
                        const float* __restrict__ wi, const float* __restrict__ wf,
                        const float* __restrict__ wo, const float* __restrict__ wz,
                        const float* __restrict__ bi, const float* __restrict__ bfv,
                        const float* __restrict__ bo, const float* __restrict__ bz,
                        const float* __restrict__ ri, const float* __restrict__ rf,
                        const float* __restrict__ ro, const float* __restrict__ rz,
                        const float* __restrict__ lni_g, const float* __restrict__ lni_b,
                        const float* __restrict__ lnf_g, const float* __restrict__ lnf_b,
                        const float* __restrict__ lno_g, const float* __restrict__ lno_b,
                        const float* __restrict__ lnz_g, const float* __restrict__ lnz_b,
                        const float* __restrict__ lnc_g, const float* __restrict__ lnc_b,
                        const float* __restrict__ lnn_g, const float* __restrict__ lnn_b,
                        const float* __restrict__ nt1, const float* __restrict__ ct1,
                        const float* __restrict__ mt1,
                        float* __restrict__ ct_sum, float* __restrict__ nt_sum,
                        float* __restrict__ obuf) {
  __shared__ float xc_s[E];
  __shared__ float xl_s[E];
  __shared__ float buf[E];
  const int row = blockIdx.x;
  const int e = threadIdx.x;
  const int d = e >> 6, o = e & 63;

  xc_s[e] = xconv[(size_t)row * E + e];
  xl_s[e] = xln[(size_t)row * E + e];
  __syncthreads();

  const int base = (d * 64 + o) * 64;
  float pi = bi[e] + ri[e];
  float pf = bfv[e] + rf[e];
  float po = bo[e] + ro[e];
  float pz = bz[e] + rz[e];
#pragma unroll 8
  for (int j = 0; j < 64; ++j) {
    const float xc = xc_s[d * 64 + j];
    const float xl = xl_s[d * 64 + j];
    pi += xc * wi[base + j];
    pf += xc * wf[base + j];
    po += xl * wo[base + j];
    pz += xl * wz[base + j];
  }

  const float li = ln_val(pi, E, true, lni_g[e], lni_b[e], buf);
  const float lf = ln_val(pf, E, true, lnf_g[e], lnf_b[e], buf);
  const float og = 1.f / (1.f + __expf(-ln_val(po, E, true, lno_g[e], lno_b[e], buf)));
  const float zg = tanhf(ln_val(pz, E, true, lnz_g[e], lnz_b[e], buf));

  const float mprev = mt1[e];
  const float m = fmaxf(lf + mprev, li);
  const float ig = __expf(li - m);
  const float fg = __expf(lf + mprev - m);

  const float ctv = fg * ct1[e] + ig * zg;
  const float ntv = fg * nt1[e] + ig;

  const float ct_ln = ln_val(ctv, E, true, lnc_g[e], lnc_b[e], buf);
  const float nt_ln = ln_val(ntv, E, true, lnn_g[e], lnn_b[e], buf);

  atomicAdd(&ct_sum[e], ct_ln);
  atomicAdd(&nt_sum[e], nt_ln);
  obuf[(size_t)row * E + e] = og;
}

// ---------------- K5a: ratio = mean(ct)/mean(nt) ----------------
__global__ void k_ratio(const float* __restrict__ ct_sum, const float* __restrict__ nt_sum,
                        float* __restrict__ ratio) {
  const int e = threadIdx.x;
  ratio[e] = ct_sum[e] / nt_sum[e];   // common 1/BT factor cancels
}

// ---------------- K5b: ht rows -> LN -> accumulate ----------------
__global__ void k_ht(const float* __restrict__ obuf, const float* __restrict__ ratio,
                     const float* __restrict__ lnh_g, const float* __restrict__ lnh_b,
                     float* __restrict__ ht_sum) {
  __shared__ float buf[E];
  const int row = blockIdx.x;
  const int e = threadIdx.x;
  const float htv = obuf[(size_t)row * E + e] * ratio[e];
  const float h = ln_val(htv, E, true, lnh_g[e], lnh_b[e], buf);
  atomicAdd(&ht_sum[e], h);
}

// ---------------- K6: gn-LN + GLU + lnout + projection (single block) ----------------
__global__ void k_tail(const float* __restrict__ ht_sum,
                       const float* __restrict__ gn_g, const float* __restrict__ gn_b,
                       const float* __restrict__ left_w, const float* __restrict__ left_b,
                       const float* __restrict__ right_w, const float* __restrict__ right_b,
                       const float* __restrict__ lnout_g, const float* __restrict__ lnout_b,
                       const float* __restrict__ proj_w, const float* __restrict__ proj_b,
                       float* __restrict__ out) {
  __shared__ float s_vec[E];
  __shared__ float glu[E43];
  __shared__ float buf[1024];
  const int t = threadIdx.x;            // 1024 threads

  float h = (t < E) ? (ht_sum[t] / (float)BT) : 0.f;
  float s = ln_val(h, E, t < E, (t < E) ? gn_g[t] : 1.f, (t < E) ? gn_b[t] : 0.f, buf);
  if (t < E) s_vec[t] = s;
  __syncthreads();

  float g = 0.f;
  if (t < E43) {
    float lv = left_b[t], rv = right_b[t];
    const float* lw = left_w + (size_t)t * E;
    const float* rw = right_w + (size_t)t * E;
#pragma unroll 8
    for (int q = 0; q < E; ++q) {
      const float sv = s_vec[q];
      lv += sv * lw[q];
      rv += sv * rw[q];
    }
    const float gelu = 0.5f * rv * (1.f + erff(rv * 0.70710678118654752f));
    g = lv * gelu;
  }
  float gl = ln_val(g, E43, t < E43, (t < E43) ? lnout_g[t] : 1.f,
                    (t < E43) ? lnout_b[t] : 0.f, buf);
  if (t < E43) glu[t] = gl;
  __syncthreads();

  if (t < E) {
    float acc = proj_b[t];
    const float* pw = proj_w + (size_t)t * E43;
#pragma unroll 2
    for (int j = 0; j < E43; ++j) acc += glu[j] * pw[j];
    out[t] = acc;
  }
}

// ---------------- launcher ----------------
extern "C" void kernel_launch(void* const* d_in, const int* in_sizes, int n_in,
                              void* d_out, int out_size, void* d_ws, size_t ws_size,
                              hipStream_t stream) {
  const float* x      = (const float*)d_in[0];
  const float* conv_w = (const float*)d_in[1];
  const float* conv_b = (const float*)d_in[2];
  const float* wi  = (const float*)d_in[3];
  const float* wf  = (const float*)d_in[4];
  const float* wo  = (const float*)d_in[5];
  const float* wz  = (const float*)d_in[6];
  const float* rwi = (const float*)d_in[7];
  const float* rwf = (const float*)d_in[8];
  const float* rwo = (const float*)d_in[9];
  const float* rwz = (const float*)d_in[10];
  const float* bi  = (const float*)d_in[11];
  const float* bfv = (const float*)d_in[12];
  const float* bo  = (const float*)d_in[13];
  const float* bz  = (const float*)d_in[14];
  const float* ln_g  = (const float*)d_in[15];
  const float* ln_b  = (const float*)d_in[16];
  const float* lni_g = (const float*)d_in[17];
  const float* lni_b = (const float*)d_in[18];
  const float* lnf_g = (const float*)d_in[19];
  const float* lnf_b = (const float*)d_in[20];
  const float* lno_g = (const float*)d_in[21];
  const float* lno_b = (const float*)d_in[22];
  const float* lnz_g = (const float*)d_in[23];
  const float* lnz_b = (const float*)d_in[24];
  const float* gn_g  = (const float*)d_in[25];
  const float* gn_b  = (const float*)d_in[26];
  const float* lnc_g = (const float*)d_in[27];
  const float* lnc_b = (const float*)d_in[28];
  const float* lnn_g = (const float*)d_in[29];
  const float* lnn_b = (const float*)d_in[30];
  const float* lnh_g = (const float*)d_in[31];
  const float* lnh_b = (const float*)d_in[32];
  const float* left_w  = (const float*)d_in[33];
  const float* left_b  = (const float*)d_in[34];
  const float* right_w = (const float*)d_in[35];
  const float* right_b = (const float*)d_in[36];
  const float* lnout_g = (const float*)d_in[37];
  const float* lnout_b = (const float*)d_in[38];
  const float* proj_w  = (const float*)d_in[39];
  const float* proj_b  = (const float*)d_in[40];
  const float* nt1 = (const float*)d_in[41];
  const float* ct1 = (const float*)d_in[42];
  const float* ht1 = (const float*)d_in[43];
  const float* mt1 = (const float*)d_in[44];
  float* out = (float*)d_out;

  char* ws = (char*)d_ws;
  float*  xln    = (float*)(ws);                                     // 8 MB
  __bf16* xbf    = (__bf16*)(ws + 8388608);                          // 4 MB
  __bf16* wpack  = (__bf16*)(ws + 12582912);                         // 32 MB
  float*  xconv  = (float*)(ws + 46137344);                          // 8 MB
  float*  obuf   = (float*)(ws + 54525952);                          // 8 MB
  float*  ri     = (float*)(ws + 62914560);
  float*  rf     = (float*)(ws + 62914560 + 2048);
  float*  ro     = (float*)(ws + 62914560 + 4096);
  float*  rz     = (float*)(ws + 62914560 + 6144);
  float*  ct_sum = (float*)(ws + 62922752);
  float*  nt_sum = (float*)(ws + 62922752 + 2048);
  float*  ht_sum = (float*)(ws + 62922752 + 4096);
  float*  ratio  = (float*)(ws + 62922752 + 6144);

  k_zero<<<6, 256, 0, stream>>>(ct_sum, 3 * E);
  k_ln_x<<<BT, E, 0, stream>>>(x, ln_g, ln_b, xln, xbf);
  k_pack_w<<<8192, 256, 0, stream>>>(conv_w, wpack);
  k_recur<<<1, E, 0, stream>>>(ht1, rwi, rwf, rwo, rwz, ri, rf, ro, rz);
  {
    dim3 grid(BT / 64, 2);
    k_conv_wmma<<<grid, 256, 0, stream>>>(xbf, wpack, conv_b, xconv);
  }
  k_gates<<<BT, E, 0, stream>>>(xconv, xln, wi, wf, wo, wz, bi, bfv, bo, bz,
                                ri, rf, ro, rz,
                                lni_g, lni_b, lnf_g, lnf_b, lno_g, lno_b,
                                lnz_g, lnz_b, lnc_g, lnc_b, lnn_g, lnn_b,
                                nt1, ct1, mt1, ct_sum, nt_sum, obuf);
  k_ratio<<<1, E, 0, stream>>>(ct_sum, nt_sum, ratio);
  k_ht<<<BT, E, 0, stream>>>(obuf, ratio, lnh_g, lnh_b, ht_sum);
  k_tail<<<1, 1024, 0, stream>>>(ht_sum, gn_g, gn_b, left_w, left_b,
                                 right_w, right_b, lnout_g, lnout_b,
                                 proj_w, proj_b, out);
}
